// CSWinBlock_12481174962258
// MI455X (gfx1250) — compile-verified, matrix operand order checked by reading, not run on hardware
//
#include <hip/hip_runtime.h>
#include <hip/hip_bf16.h>

// ---------------------------------------------------------------------------
// CDNA5 (gfx1250) CSWin block. GEMMs + attention on v_wmma_f32_16x16x32_f16.
// B operands pre-packed into fragment-native layout -> per-lane contiguous
// 32B loads (global_load_b128 x2). A operands via row-major LDS with
// vectorized ds_load_b128 fragment assembly.
// ---------------------------------------------------------------------------

typedef __attribute__((ext_vector_type(16))) _Float16 v16h;
typedef __attribute__((ext_vector_type(8)))  _Float16 v8h;
typedef __attribute__((ext_vector_type(8)))  float    v8f;

#define WMMA_F16(a, b, c) \
  __builtin_amdgcn_wmma_f32_16x16x32_f16(false, (a), false, (b), (short)0, (c), false, false)

__device__ __forceinline__ float gelu_exact(float x) {
  return 0.5f * x * (1.0f + erff(x * 0.70710678118654752f));
}

// A fragment (16x32 f16) from row-major LDS tile, row stride `ld` halves.
// Lane l: row = l&15, k-runs [kh, kh+8) and [kh+16, kh+24), kh = 8*(l>>4).
// Two contiguous 16B runs -> 2x ds_load_b128.
__device__ __forceinline__ v16h frag_a_rowmajor(const _Float16* base, int ld, int lane) {
  const int row = lane & 15;
  const int kh  = (lane >> 4) * 8;
  const v8h lo = *(const v8h*)(base + row * ld + kh);
  const v8h hi = *(const v8h*)(base + row * ld + kh + 16);
  return __builtin_shufflevector(lo, hi, 0, 1, 2, 3, 4, 5, 6, 7,
                                 8, 9, 10, 11, 12, 13, 14, 15);
}

// B fragment (32x16 f16) from fragment-packed global memory:
// frag = 512 halves, lane l owns halves [l*16, l*16+16) -> 32B contiguous.
__device__ __forceinline__ v16h frag_b_packed(const _Float16* frag_base, int lane) {
  return *(const v16h*)(frag_base + lane * 16);
}

// Packed-slot index for B element (k in [0,32), n in [0,16)):
//   lane = (n&15) + 16*((k>>3)&1),  elem = (k&7) + 8*(k>>4)
__device__ __forceinline__ int b_slot(int k, int n) {
  return ((n & 15) + 16 * ((k >> 3) & 1)) * 16 + (k & 7) + 8 * (k >> 4);
}

// half-wave (16 lane) reductions over score C-fragment columns
__device__ __forceinline__ float half16_max(float v) {
#pragma unroll
  for (int m = 1; m <= 8; m <<= 1) v = fmaxf(v, __shfl_xor(v, m, 32));
  return v;
}
__device__ __forceinline__ float half16_sum(float v) {
#pragma unroll
  for (int m = 1; m <= 8; m <<= 1) v += __shfl_xor(v, m, 32);
  return v;
}

// window position -> L (branch 0: 64x8 column strips, branch 1: 8x64 rows)
__device__ __forceinline__ int pos_to_L(int branch, int sub, int p) {
  if (branch == 0) { int h = p >> 3, win = p & 7; return h * 64 + sub * 8 + win; }
  else             { int hin = p >> 6, w = p & 63; return (sub * 8 + hin) * 64 + w; }
}

// ---------------------------------------------------------------------------
// Weight pack: f32 [K,N] -> f16 fragment-packed [N/16][K/32][32][16]
// ---------------------------------------------------------------------------
__global__ __launch_bounds__(256) void wpack_kernel(const float* __restrict__ w,
                                                    _Float16* __restrict__ out,
                                                    int K, int N) {
  const int i = blockIdx.x * 256 + threadIdx.x;
  if (i >= K * N) return;
  const int k = i / N, n = i % N;
  const int KT = K >> 5;
  const size_t dst = (size_t)((n >> 4) * KT + (k >> 5)) * 512 + b_slot(k & 31, n & 15);
  out[dst] = (_Float16)w[i];
}

// ---------------------------------------------------------------------------
// K pack: qkv f16 -> per (branch,window,head) B-frag tiles for Q@K^T.
// B matrix is [k=dim(32 padded)][n=key]; frag f = key-tile of 16.
// Layout: [br][wi][hd][32 frags][512 halves]. One thread per key: the three
// live 8-dim runs + one zero run are each a contiguous 16B store.
// ---------------------------------------------------------------------------
__global__ __launch_bounds__(256) void kpack_kernel(const _Float16* __restrict__ qkv,
                                                    _Float16* __restrict__ kp) {
  const int idx = blockIdx.x * 256 + threadIdx.x;  // 2*64*4*512
  const int key = idx & 511;
  const int hd  = (idx >> 9) & 3;
  const int wi  = (idx >> 11) & 63;
  const int br  = (idx >> 17) & 1;
  const int b   = wi >> 3, sub = wi & 7;
  const int cb  = br * 96 + hd * 24;
  const int L   = pos_to_L(br, sub, key);
  const _Float16* src = qkv + ((size_t)(b * 4096 + L)) * 576 + 192 + cb;

  _Float16* base = kp + ((size_t)(((br * 64 + wi) * 4 + hd) * 32 + (key >> 4))) * 512;
  const int klane = key & 15;

  v8h d0, d1, d2, z = {};
#pragma unroll
  for (int d = 0; d < 8; ++d) { d0[d] = src[d]; d1[d] = src[8 + d]; d2[d] = src[16 + d]; }
  *(v8h*)(base + klane * 16)            = d0;  // dims 0..7   (lane=klane,    e0..7)
  *(v8h*)(base + (klane + 16) * 16)     = d1;  // dims 8..15  (lane=klane+16, e0..7)
  *(v8h*)(base + klane * 16 + 8)        = d2;  // dims 16..23 (lane=klane,    e8..15)
  *(v8h*)(base + (klane + 16) * 16 + 8) = z;   // dims 24..31 zero pad
}

// ---------------------------------------------------------------------------
// V pack: qkv f16 -> B-frag tiles for P@V. B matrix is [k=key][n=dim(32 pad)].
// Per 64-key tile kt: frags (ks2 in {0,1}) x (nt2 in {0,1}).
// Layout: [br][wi][hd][32 frags][512]. One thread per (dim, 8-key run):
// gathers 8 halves, stores one contiguous 16B chunk.
// ---------------------------------------------------------------------------
__global__ __launch_bounds__(256) void vpack_kernel(const _Float16* __restrict__ qkv,
                                                    _Float16* __restrict__ vp) {
  const int idx = blockIdx.x * 256 + threadIdx.x;  // 2*64*4*32*64 = 2^20
  const int run = idx & 63;          // 8-key run within 512
  const int dim = (idx >> 6) & 31;
  const int hd  = (idx >> 11) & 3;
  const int wi  = (idx >> 13) & 63;
  const int br  = (idx >> 19) & 1;
  const int b   = wi >> 3, sub = wi & 7;
  const int cb  = br * 96 + hd * 24;

  const int key0 = run * 8;
  const int kt   = key0 >> 6;
  const int kk0  = key0 & 63;        // within tile
  const int ks2  = kk0 >> 5;
  const int kl0  = kk0 & 31;         // multiple of 8
  const int nt2  = dim >> 4;
  const int frag = kt * 4 + ks2 * 2 + nt2;
  const int lane = (dim & 15) + 16 * ((kl0 >> 3) & 1);
  const int e0   = 8 * (kl0 >> 4);

  v8h t = {};
  if (dim < 24) {
#pragma unroll
    for (int j = 0; j < 8; ++j) {
      const int L = pos_to_L(br, sub, key0 + j);
      t[j] = qkv[((size_t)(b * 4096 + L)) * 576 + 384 + cb + dim];
    }
  }
  _Float16* base = vp + ((size_t)(((br * 64 + wi) * 4 + hd) * 32 + frag)) * 512;
  *(v8h*)(base + lane * 16 + e0) = t;
}

// ---------------------------------------------------------------------------
// LayerNorm: one wave per row of C=192 (6 floats/lane, wave32 shuffles)
// ---------------------------------------------------------------------------
__global__ __launch_bounds__(256) void ln_kernel(const float* __restrict__ x,
                                                 const float* __restrict__ g,
                                                 const float* __restrict__ b,
                                                 _Float16* __restrict__ out) {
  const int C    = 192;
  const int lane = threadIdx.x & 31;
  const int wave = threadIdx.x >> 5;
  const int row  = blockIdx.x * 8 + wave;
  const float* xr = x + (size_t)row * C;

  float vals[6];
  float s = 0.f;
#pragma unroll
  for (int i = 0; i < 6; ++i) { vals[i] = xr[lane + 32 * i]; s += vals[i]; }
#pragma unroll
  for (int m = 16; m >= 1; m >>= 1) s += __shfl_xor(s, m, 32);
  const float mean = s * (1.0f / 192.0f);

  float vs = 0.f;
#pragma unroll
  for (int i = 0; i < 6; ++i) { float d = vals[i] - mean; vs += d * d; }
#pragma unroll
  for (int m = 16; m >= 1; m >>= 1) vs += __shfl_xor(vs, m, 32);
  const float inv = rsqrtf(vs * (1.0f / 192.0f) + 1e-5f);

#pragma unroll
  for (int i = 0; i < 6; ++i) {
    const int c = lane + 32 * i;
    out[(size_t)row * C + c] = (_Float16)((vals[i] - mean) * inv * g[c] + b[c]);
  }
}

// ---------------------------------------------------------------------------
// WMMA GEMM: out[M,N] = A[M,K] @ Bpack (+epilogue). A via LDS (row-major,
// vector ds); B direct from fragment-packed global (L2-hot weights).
// 128x64 block, BK=32, 8 waves (4x2), wave tile 32x32.
// EPI: 0 store f16; 2 +bias,GELU f16; 3 +bias,+f32 residual, store f32
// ---------------------------------------------------------------------------
#define GBM 128
#define GBN 64
#define GBK 32

template <int EPI>
__global__ __launch_bounds__(256) void gemm_wmma_kernel(const _Float16* __restrict__ A,
                                                        const _Float16* __restrict__ Bpack,
                                                        const float* __restrict__ bias,
                                                        const float* __restrict__ resid,
                                                        void* __restrict__ outv,
                                                        int M, int N, int K) {
  __shared__ _Float16 As[GBM][GBK + 8];

  const int tid  = threadIdx.x;
  const int lane = tid & 31;
  const int wave = tid >> 5;
  const int wm   = wave & 3;
  const int wn   = wave >> 2;
  const int bm   = blockIdx.y * GBM;
  const int bn   = blockIdx.x * GBN;
  const int KT   = K >> 5;

  v8f acc[2][2] = {};

  const int arow = tid >> 1;
  const int akc  = (tid & 1) * 16;

  // per-wave fixed N tiles in the packed-B layout
  const _Float16* bbase[2];
#pragma unroll
  for (int j = 0; j < 2; ++j)
    bbase[j] = Bpack + (size_t)(((bn >> 4) + wn * 2 + j) * KT) * 512;

  for (int k0 = 0; k0 < K; k0 += GBK) {
    {
      const _Float16* src = A + (size_t)(bm + arow) * K + (k0 + akc);
      *(uint4*)&As[arow][akc]     = *(const uint4*)(src);
      *(uint4*)&As[arow][akc + 8] = *(const uint4*)(src + 8);
      if (k0 + GBK < K) __builtin_prefetch(src + GBK, 0, 1);
    }
    __syncthreads();

    v16h af[2], bf[2];
#pragma unroll
    for (int i = 0; i < 2; ++i) af[i] = frag_a_rowmajor(&As[wm * 32 + i * 16][0], GBK + 8, lane);
#pragma unroll
    for (int j = 0; j < 2; ++j) bf[j] = frag_b_packed(bbase[j] + (size_t)(k0 >> 5) * 512, lane);
#pragma unroll
    for (int i = 0; i < 2; ++i)
#pragma unroll
      for (int j = 0; j < 2; ++j) acc[i][j] = WMMA_F16(af[i], bf[j], acc[i][j]);

    __syncthreads();
  }

  const int half  = lane >> 4;
  const int col_l = lane & 15;
#pragma unroll
  for (int i = 0; i < 2; ++i) {
#pragma unroll
    for (int j = 0; j < 2; ++j) {
      const int c = bn + wn * 32 + j * 16 + col_l;
      float bv = 0.f;
      if (EPI >= 1) bv = bias[c];
#pragma unroll
      for (int r = 0; r < 8; ++r) {
        const int    rr  = bm + wm * 32 + i * 16 + r + 8 * half;
        const size_t idx = (size_t)rr * N + c;
        float v = acc[i][j][r] + bv;
        if (EPI == 2) v = gelu_exact(v);
        if (EPI == 3) {
          ((float*)outv)[idx] = v + resid[idx];
        } else {
          ((_Float16*)outv)[idx] = (_Float16)v;
        }
      }
    }
  }
}

// ---------------------------------------------------------------------------
// CSWin window attention: WMMA + two-pass streaming softmax.
// One block (4 waves) per (window, head, 64-query tile). S=512, d=24->32.
// K/V fragments come straight from packed global; only Q and P touch LDS.
// ---------------------------------------------------------------------------
__global__ __launch_bounds__(128) void win_attn_kernel(const _Float16* __restrict__ qkv,
                                                       const _Float16* __restrict__ kp,
                                                       const _Float16* __restrict__ vp,
                                                       _Float16* __restrict__ attn,
                                                       int branch) {
  __shared__ _Float16 Qs[64][40];      // queries x padded dims (row-major)
  __shared__ _Float16 Ps[4][16][72];   // per-wave P tile 16x64 (row-major)

  const int bid = blockIdx.x;          // 64 win * 4 heads * 8 qtiles
  const int qt  = bid & 7;
  const int hd  = (bid >> 3) & 3;
  const int wi  = bid >> 5;
  const int b   = wi >> 3;
  const int sub = wi & 7;
  const int cb  = branch * 96 + hd * 24;

  const int tid  = threadIdx.x;
  const int lane = tid & 31;
  const int wv   = tid >> 5;
  const float scale = 0.2041241452319315f;  // 24^-0.5

  const _Float16* kwin = kp + ((size_t)(((branch * 64 + wi) * 4 + hd) * 32)) * 512;
  const _Float16* vwin = vp + ((size_t)(((branch * 64 + wi) * 4 + hd) * 32)) * 512;

  // ---- stage Q (64 x 24 -> 64 x 32, pre-scaled), vector LDS stores ----
  {
    const int r     = tid >> 1;
    const int dbase = (tid & 1) * 16;
    const int L     = pos_to_L(branch, sub, qt * 64 + r);
    const _Float16* src = qkv + ((size_t)(b * 4096 + L)) * 576 + cb;
    v8h q0 = {}, q1 = {};
#pragma unroll
    for (int d2 = 0; d2 < 8; ++d2) {
      const int dim = dbase + d2;
      if (dim < 24) q0[d2] = (_Float16)((float)src[dim] * scale);
      const int dim2 = dbase + 8 + d2;
      if (dim2 < 24) q1[d2] = (_Float16)((float)src[dim2] * scale);
    }
    *(v8h*)&Qs[r][dbase]     = q0;
    *(v8h*)&Qs[r][dbase + 8] = q1;
  }
  __syncthreads();

  const v16h qf = frag_a_rowmajor(&Qs[wv * 16][0], 40, lane);

  float mrow[8], lrow[8];
#pragma unroll
  for (int r = 0; r < 8; ++r) { mrow[r] = -1e30f; lrow[r] = 0.f; }

  // ---- pass 1: row max ----
  for (int kt = 0; kt < 8; ++kt) {
    float pm[8];
#pragma unroll
    for (int r = 0; r < 8; ++r) pm[r] = -1e30f;
#pragma unroll
    for (int nt = 0; nt < 4; ++nt) {
      const v16h bf = frag_b_packed(kwin + (size_t)(kt * 4 + nt) * 512, lane);
      v8f s = {};
      s = WMMA_F16(qf, bf, s);
#pragma unroll
      for (int r = 0; r < 8; ++r) pm[r] = fmaxf(pm[r], s[r]);
    }
#pragma unroll
    for (int r = 0; r < 8; ++r) mrow[r] = fmaxf(mrow[r], half16_max(pm[r]));
  }

  // ---- pass 2: recompute, exp, O = P @ V ----
  v8f Of[2] = {};
  for (int kt = 0; kt < 8; ++kt) {
    float psum[8];
#pragma unroll
    for (int r = 0; r < 8; ++r) psum[r] = 0.f;
#pragma unroll
    for (int nt = 0; nt < 4; ++nt) {
      const v16h bf = frag_b_packed(kwin + (size_t)(kt * 4 + nt) * 512, lane);
      v8f s = {};
      s = WMMA_F16(qf, bf, s);
#pragma unroll
      for (int r = 0; r < 8; ++r) {
        const float p = expf(s[r] - mrow[r]);
        psum[r] += p;
        Ps[wv][r + 8 * (lane >> 4)][nt * 16 + (lane & 15)] = (_Float16)p;
      }
    }
#pragma unroll
    for (int r = 0; r < 8; ++r) lrow[r] += half16_sum(psum[r]);
    __syncthreads();
#pragma unroll
    for (int ks2 = 0; ks2 < 2; ++ks2) {
      const v16h pf = frag_a_rowmajor(&Ps[wv][0][ks2 * 32], 72, lane);
#pragma unroll
      for (int nt2 = 0; nt2 < 2; ++nt2) {
        const v16h vf = frag_b_packed(vwin + (size_t)(kt * 4 + ks2 * 2 + nt2) * 512, lane);
        Of[nt2] = WMMA_F16(pf, vf, Of[nt2]);
      }
    }
    __syncthreads();
  }

  // ---- normalize, scatter back to (B, L, 192) ----
#pragma unroll
  for (int nt2 = 0; nt2 < 2; ++nt2) {
    const int dim = nt2 * 16 + (lane & 15);
    if (dim < 24) {
#pragma unroll
      for (int r = 0; r < 8; ++r) {
        const int p = qt * 64 + wv * 16 + r + 8 * (lane >> 4);
        const int L = pos_to_L(branch, sub, p);
        attn[((size_t)(b * 4096 + L)) * 192 + cb + dim] = (_Float16)(Of[nt2][r] / lrow[r]);
      }
    }
  }
}

// ---------------------------------------------------------------------------
// Depthwise 3x3 conv (+bias, GELU), NHWC, vectorized over 8-channel groups:
// every tap is a 16B global_load_b128, store is a 16B global_store_b128.
// ---------------------------------------------------------------------------
__global__ __launch_bounds__(256) void dwconv_gelu_kernel(const _Float16* __restrict__ t,
                                                          const float* __restrict__ w,
                                                          const float* __restrict__ bias,
                                                          _Float16* __restrict__ out) {
  const int H = 64, W = 64, Ch = 768, G = Ch / 8;  // 96 channel groups
  const size_t i = (size_t)blockIdx.x * 256 + threadIdx.x;  // B*H*W*G
  const int g  = (int)(i % G);
  size_t sp    = i / G;
  const int wx = (int)(sp % W); sp /= W;
  const int hy = (int)(sp % H);
  const int b  = (int)(sp / H);
  const int c0 = g * 8;

  float acc[8];
#pragma unroll
  for (int u = 0; u < 8; ++u) acc[u] = bias[c0 + u];

#pragma unroll
  for (int dy = -1; dy <= 1; ++dy) {
#pragma unroll
    for (int dx = -1; dx <= 1; ++dx) {
      const int yy = hy + dy, xx = wx + dx;
      if (yy >= 0 && yy < H && xx >= 0 && xx < W) {
        const v8h tv = *(const v8h*)(t + (((size_t)b * H + yy) * W + xx) * Ch + c0);
#pragma unroll
        for (int u = 0; u < 8; ++u)
          acc[u] += (float)tv[u] * w[(c0 + u) * 9 + (dy + 1) * 3 + (dx + 1)];
      }
    }
  }

  v8h o;
#pragma unroll
  for (int u = 0; u < 8; ++u) o[u] = (_Float16)gelu_exact(acc[u]);
  *(v8h*)(out + (((size_t)b * H + hy) * W + wx) * Ch + c0) = o;
}

// ---------------------------------------------------------------------------
// Host-side orchestration
// ---------------------------------------------------------------------------
extern "C" void kernel_launch(void* const* d_in, const int* in_sizes, int n_in,
                              void* d_out, int out_size, void* d_ws, size_t ws_size,
                              hipStream_t stream) {
  const float* x      = (const float*)d_in[0];
  const float* qkv_w  = (const float*)d_in[1];
  const float* proj_w = (const float*)d_in[2];
  const float* proj_b = (const float*)d_in[3];
  const float* n1g    = (const float*)d_in[4];
  const float* n1b    = (const float*)d_in[5];
  const float* n2g    = (const float*)d_in[6];
  const float* n2b    = (const float*)d_in[7];
  const float* lin1_w = (const float*)d_in[8];
  const float* lin1_b = (const float*)d_in[9];
  const float* dw_w   = (const float*)d_in[10];
  const float* dw_b   = (const float*)d_in[11];
  const float* lin2_w = (const float*)d_in[12];
  const float* lin2_b = (const float*)d_in[13];
  (void)in_sizes; (void)n_in; (void)out_size; (void)ws_size;

  const int M = 32768;

  char* ws = (char*)d_ws;
  _Float16* w_qkv16  = (_Float16*)(ws + 0);
  _Float16* w_proj16 = w_qkv16  + 192 * 576;
  _Float16* w_lin116 = w_proj16 + 192 * 192;
  _Float16* w_lin216 = w_lin116 + 192 * 768;                         // < 1 MB
  _Float16* xn16     = (_Float16*)(ws + (size_t)(1)  * (1u << 20));  // 12.6 MB
  _Float16* qkv16    = (_Float16*)(ws + (size_t)(14) * (1u << 20));  // 37.7 MB
  _Float16* attn16   = xn16;                                         // reuse
  _Float16* kpack    = (_Float16*)(ws + (size_t)(52) * (1u << 20));  // 16 MB, dies before y32
  _Float16* vpack    = (_Float16*)(ws + (size_t)(68) * (1u << 20));  // 16 MB, dies before y32/yn
  float*    y32      = (float*)   (ws + (size_t)(52) * (1u << 20));  // 25.2 MB (after attention)
  _Float16* yn16     = (_Float16*)(ws + (size_t)(77) * (1u << 20));  // 12.6 MB
  _Float16* t16      = (_Float16*)(ws + (size_t)(1)  * (1u << 20));  // reuse xn+qkv
  _Float16* tdw16    = (_Float16*)(ws + (size_t)(77) * (1u << 20));  // reuse yn

  // 1) weights -> fragment-packed f16
  wpack_kernel<<<(192 * 576 + 255) / 256, 256, 0, stream>>>(qkv_w,  w_qkv16,  192, 576);
  wpack_kernel<<<(192 * 192 + 255) / 256, 256, 0, stream>>>(proj_w, w_proj16, 192, 192);
  wpack_kernel<<<(192 * 768 + 255) / 256, 256, 0, stream>>>(lin1_w, w_lin116, 192, 768);
  wpack_kernel<<<(768 * 192 + 255) / 256, 256, 0, stream>>>(lin2_w, w_lin216, 768, 192);

  // 2) LN1
  ln_kernel<<<M / 8, 256, 0, stream>>>(x, n1g, n1b, xn16);

  // 3) QKV GEMM
  gemm_wmma_kernel<0><<<dim3(576 / GBN, M / GBM), 256, 0, stream>>>(
      xn16, w_qkv16, nullptr, nullptr, qkv16, M, 576, 192);

  // 4) pack K/V into fragment-native windowed layout, then attention
  kpack_kernel<<<(2 * 64 * 4 * 512) / 256, 256, 0, stream>>>(qkv16, kpack);
  vpack_kernel<<<(2 * 64 * 4 * 32 * 64) / 256, 256, 0, stream>>>(qkv16, vpack);
  win_attn_kernel<<<2048, 128, 0, stream>>>(qkv16, kpack, vpack, attn16, 0);
  win_attn_kernel<<<2048, 128, 0, stream>>>(qkv16, kpack, vpack, attn16, 1);

  // 5) proj + bias + residual(x) -> y (f32)
  gemm_wmma_kernel<3><<<dim3(192 / GBN, M / GBM), 256, 0, stream>>>(
      attn16, w_proj16, proj_b, x, y32, M, 192, 192);

  // 6) LN2
  ln_kernel<<<M / 8, 256, 0, stream>>>(y32, n2g, n2b, yn16);

  // 7) lin1 + bias + GELU -> t (f16)
  gemm_wmma_kernel<2><<<dim3(768 / GBN, M / GBM), 256, 0, stream>>>(
      yn16, w_lin116, lin1_b, nullptr, t16, M, 768, 192);

  // 8) depthwise 3x3 conv + bias + GELU (vectorized, b128 taps)
  dwconv_gelu_kernel<<<(8 * 64 * 64 * 96) / 256, 256, 0, stream>>>(t16, dw_w, dw_b, tdw16);

  // 9) lin2 + bias + residual(y) -> d_out (f32)
  gemm_wmma_kernel<3><<<dim3(192 / GBN, M / GBM), 256, 0, stream>>>(
      tdw16, w_lin216, lin2_b, y32, (float*)d_out, M, 192, 768);
}
